// TimeMoeSparseExpertsLayer_53584011985414
// MI455X (gfx1250) — matrix operand order repaired and checked by reading, not compiled, not run on hardware
//
#include <hip/hip_runtime.h>
#include <hip/hip_bf16.h>

// ---------------- problem constants ----------------
#define T_TOK   16384   // B*S
#define DMODEL  1024
#define NEXP    8
#define TOPK    2
#define FE      2048    // per-expert intermediate
#define DFF     4096    // shared expert intermediate
#define MAX_SLOTS 33024 // T*TOPK + NEXP*31, rounded to multiple of 32

// ---------------- CDNA5 WMMA types ----------------
typedef __bf16 bf16x16 __attribute__((ext_vector_type(16)));
typedef float  floatx8 __attribute__((ext_vector_type(8)));
typedef unsigned int u32x4 __attribute__((ext_vector_type(4)));

union FragU { bf16x16 v; u32x4 q[2]; };

__device__ __forceinline__ unsigned short f2bf(float f) {
    unsigned u = __float_as_uint(f);
    u += 0x7FFFu + ((u >> 16) & 1u);     // round-to-nearest-even
    return (unsigned short)(u >> 16);
}

// ---------------- fp32 -> bf16 conversion ----------------
__global__ void cvt_kernel(const float* __restrict__ s,
                           unsigned short* __restrict__ d, size_t n) {
    size_t i = ((size_t)blockIdx.x * blockDim.x + threadIdx.x) * 4;
    if (i + 3 >= n) return;   // all sizes are multiples of 4
    float4 f = *(const float4*)(s + i);
    unsigned lo = (unsigned)f2bf(f.x) | ((unsigned)f2bf(f.y) << 16);
    unsigned hi = (unsigned)f2bf(f.z) | ((unsigned)f2bf(f.w) << 16);
    *(uint2*)(d + i) = make_uint2(lo, hi);
}

// ---------------- init scratch state ----------------
__global__ void init_kernel(int* __restrict__ slot_tok,
                            float* __restrict__ slot_w,
                            int* __restrict__ cnt) {
    int i = blockIdx.x * blockDim.x + threadIdx.x;
    if (i < MAX_SLOTS) { slot_tok[i] = -1; slot_w[i] = 0.0f; }
    if (i < NEXP) cnt[i] = 0;
}

// ---------------- router: logits, softmax, top-2, shared gate ----------------
__global__ void __launch_bounds__(256) router_kernel(
    const float* __restrict__ x, const float* __restrict__ gate_w,
    const float* __restrict__ sgate_w, float* __restrict__ logits,
    int* __restrict__ topk_idx, float* __restrict__ topk_w,
    float* __restrict__ sg, int* __restrict__ cnt) {
    int lane = threadIdx.x & 31;
    int wave = threadIdx.x >> 5;
    int t = blockIdx.x * 8 + wave;
    const float* xr = x + (size_t)t * DMODEL;
    float acc[9];
#pragma unroll
    for (int i = 0; i < 9; ++i) acc[i] = 0.0f;
    for (int dd = lane; dd < DMODEL; dd += 32) {
        float xv = xr[dd];
#pragma unroll
        for (int e = 0; e < NEXP; ++e) acc[e] += xv * gate_w[e * DMODEL + dd];
        acc[8] += xv * sgate_w[dd];
    }
#pragma unroll
    for (int i = 0; i < 9; ++i)
        for (int m = 16; m > 0; m >>= 1) acc[i] += __shfl_xor(acc[i], m, 32);
    if (lane < NEXP) logits[(size_t)t * NEXP + lane] = acc[lane];
    if (lane == 0) {
        float mx = acc[0];
#pragma unroll
        for (int e = 1; e < NEXP; ++e) mx = fmaxf(mx, acc[e]);
        float p[NEXP], s = 0.0f;
#pragma unroll
        for (int e = 0; e < NEXP; ++e) { p[e] = __expf(acc[e] - mx); s += p[e]; }
        float inv = 1.0f / s;
#pragma unroll
        for (int e = 0; e < NEXP; ++e) p[e] *= inv;
        int i1 = 0;
#pragma unroll
        for (int e = 1; e < NEXP; ++e) if (p[e] > p[i1]) i1 = e;   // first max on ties
        int i2 = (i1 == 0) ? 1 : 0;
#pragma unroll
        for (int e = 0; e < NEXP; ++e)
            if (e != i1 && p[e] > p[i2]) i2 = e;
        topk_idx[t * 2 + 0] = i1; topk_idx[t * 2 + 1] = i2;
        topk_w[t * 2 + 0] = p[i1]; topk_w[t * 2 + 1] = p[i2];
        atomicAdd(&cnt[i1], 1); atomicAdd(&cnt[i2], 1);
        sg[t] = 1.0f / (1.0f + __expf(-acc[8]));
    }
}

// ---------------- padded (to 32) prefix sums ----------------
__global__ void offsets_kernel(const int* __restrict__ cnt,
                               int* __restrict__ off, int* __restrict__ cur) {
    if (threadIdx.x == 0 && blockIdx.x == 0) {
        int acc = 0;
        for (int e = 0; e < NEXP; ++e) {
            off[e] = acc; cur[e] = acc;
            acc += (cnt[e] + 31) & ~31;
        }
        off[NEXP] = acc;
    }
}

// ---------------- build gathered token lists ----------------
__global__ void fill_kernel(const int* __restrict__ ti, const float* __restrict__ tw,
                            int* __restrict__ cur, int* __restrict__ slot_tok,
                            float* __restrict__ slot_w) {
    int t = blockIdx.x * blockDim.x + threadIdx.x;
    if (t >= T_TOK) return;
#pragma unroll
    for (int k = 0; k < TOPK; ++k) {
        int e = ti[t * 2 + k];
        int pos = atomicAdd(&cur[e], 1);
        slot_tok[pos] = t;
        slot_w[pos] = tw[t * 2 + k];
    }
}

// ---------------- fc1: h = silu(x Wg^T) * (x Wu^T) * w_row ----------------
// wave tile: M=32 x N=64, block = 8 waves -> N=512; K-loop over DMODEL in
// steps of 32, software-pipelined 2-deep (ping-pong register buffers).
__global__ void __launch_bounds__(256) fc1_kernel(
    const unsigned short* __restrict__ xb,
    const unsigned short* __restrict__ wg,
    const unsigned short* __restrict__ wu,
    unsigned short* __restrict__ hbuf,
    const int* __restrict__ slot_tok,
    const float* __restrict__ slot_w,   // routed weight, or sigmoid shared-gate
    const int* __restrict__ offsets,
    int F, int shared_mode) {
    const int Dk = DMODEL;              // multiple of 64
    int s0 = blockIdx.x * 32;
    int e = 0;
    if (!shared_mode) {
        if (s0 >= offsets[NEXP]) return;
        while (e < NEXP - 1 && s0 >= offsets[e + 1]) ++e;
    }
    int lane = threadIdx.x & 31, wave = threadIdx.x >> 5;
    int col = lane & 15, khalf = lane >> 4;
    int n0 = blockIdx.y * 512 + wave * 64;

    const unsigned short* rowA[2];
#pragma unroll
    for (int mi = 0; mi < 2; ++mi) {
        int slot = s0 + mi * 16 + col;
        int tok = shared_mode ? slot : slot_tok[slot];
        if (tok < 0) tok = 0;               // padded slot: weight==0 anyway
        rowA[mi] = xb + (size_t)tok * Dk;
    }
    const unsigned short* rowG[4];
    const unsigned short* rowU[4];
#pragma unroll
    for (int nj = 0; nj < 4; ++nj) {
        size_t wr = (size_t)e * F + (size_t)(n0 + nj * 16 + col);
        rowG[nj] = wg + wr * Dk;
        rowU[nj] = wu + wr * Dk;
    }

    floatx8 cg[2][4], cu[2][4];
#pragma unroll
    for (int mi = 0; mi < 2; ++mi)
#pragma unroll
        for (int nj = 0; nj < 4; ++nj)
#pragma unroll
            for (int r = 0; r < 8; ++r) { cg[mi][nj][r] = 0.0f; cu[mi][nj][r] = 0.0f; }

    FragU a0[2], a1[2], bg0[4], bg1[4], bu0[4], bu1[4];

#define FC1_LOAD(AA, BG, BU, kk) do {                                       \
        _Pragma("unroll")                                                   \
        for (int mi = 0; mi < 2; ++mi) {                                    \
            const unsigned short* p = rowA[mi] + (kk) + khalf * 16;         \
            AA[mi].q[0] = *(const u32x4*)p;                                 \
            AA[mi].q[1] = *(const u32x4*)(p + 8);                           \
        }                                                                   \
        _Pragma("unroll")                                                   \
        for (int nj = 0; nj < 4; ++nj) {                                    \
            const unsigned short* pg = rowG[nj] + (kk) + khalf * 16;        \
            BG[nj].q[0] = *(const u32x4*)pg;                                \
            BG[nj].q[1] = *(const u32x4*)(pg + 8);                          \
            const unsigned short* pu = rowU[nj] + (kk) + khalf * 16;        \
            BU[nj].q[0] = *(const u32x4*)pu;                                \
            BU[nj].q[1] = *(const u32x4*)(pu + 8);                          \
        }                                                                   \
    } while (0)

#define FC1_MMA(AA, BG, BU) do {                                            \
        _Pragma("unroll")                                                   \
        for (int nj = 0; nj < 4; ++nj) {                                    \
            _Pragma("unroll")                                               \
            for (int mi = 0; mi < 2; ++mi) {                                \
                cg[mi][nj] = __builtin_amdgcn_wmma_f32_16x16x32_bf16(       \
                    false, AA[mi].v, false, BG[nj].v, (short)0,             \
                    cg[mi][nj], false, false);                              \
                cu[mi][nj] = __builtin_amdgcn_wmma_f32_16x16x32_bf16(       \
                    false, AA[mi].v, false, BU[nj].v, (short)0,             \
                    cu[mi][nj], false, false);                              \
            }                                                               \
        }                                                                   \
    } while (0)

    FC1_LOAD(a0, bg0, bu0, 0);
    for (int k0 = 0; k0 < Dk; k0 += 64) {
        __builtin_prefetch(rowG[0] + k0 + 64, 0, 1);
        __builtin_prefetch(rowU[0] + k0 + 64, 0, 1);
        FC1_LOAD(a1, bg1, bu1, k0 + 32);          // next slab in flight
        FC1_MMA(a0, bg0, bu0);                    // consume current slab
        if (k0 + 64 < Dk) FC1_LOAD(a0, bg0, bu0, k0 + 64);
        FC1_MMA(a1, bg1, bu1);
    }
#undef FC1_LOAD
#undef FC1_MMA

#pragma unroll
    for (int mi = 0; mi < 2; ++mi) {
        int mbase = s0 + mi * 16 + khalf * 8;   // C-frag rows for this lane
        float wrow[8];
#pragma unroll
        for (int r = 0; r < 8; ++r) wrow[r] = slot_w[mbase + r];
#pragma unroll
        for (int nj = 0; nj < 4; ++nj) {
            int n = n0 + nj * 16 + col;
#pragma unroll
            for (int r = 0; r < 8; ++r) {
                float g = cg[mi][nj][r];
                float u = cu[mi][nj][r];
                float h = (g / (1.0f + __expf(-g))) * u * wrow[r];
                hbuf[(size_t)(mbase + r) * F + n] = f2bf(h);
            }
        }
    }
}

// ---------------- fc2: y = h Wd^T ; shared -> store, routed -> atomic add ----
__global__ void __launch_bounds__(256) fc2_kernel(
    const unsigned short* __restrict__ hbuf,
    const unsigned short* __restrict__ wd,
    float* __restrict__ out,
    const int* __restrict__ slot_tok,
    const int* __restrict__ offsets,
    int F, int shared_mode) {
    const int Dout = DMODEL;
    int s0 = blockIdx.x * 32;
    int e = 0;
    if (!shared_mode) {
        if (s0 >= offsets[NEXP]) return;
        while (e < NEXP - 1 && s0 >= offsets[e + 1]) ++e;
    }
    int lane = threadIdx.x & 31, wave = threadIdx.x >> 5;
    int col = lane & 15, khalf = lane >> 4;
    int n0 = blockIdx.y * 512 + wave * 64;

    const unsigned short* rowA[2];
#pragma unroll
    for (int mi = 0; mi < 2; ++mi)
        rowA[mi] = hbuf + (size_t)(s0 + mi * 16 + col) * F;
    const unsigned short* rowB[4];
#pragma unroll
    for (int nj = 0; nj < 4; ++nj)
        rowB[nj] = wd + ((size_t)e * Dout + (size_t)(n0 + nj * 16 + col)) * F;

    floatx8 c[2][4];
#pragma unroll
    for (int mi = 0; mi < 2; ++mi)
#pragma unroll
        for (int nj = 0; nj < 4; ++nj)
#pragma unroll
            for (int r = 0; r < 8; ++r) c[mi][nj][r] = 0.0f;

    FragU a0[2], a1[2], b0[4], b1[4];

#define FC2_LOAD(AA, BB, kk) do {                                           \
        _Pragma("unroll")                                                   \
        for (int mi = 0; mi < 2; ++mi) {                                    \
            const unsigned short* p = rowA[mi] + (kk) + khalf * 16;         \
            AA[mi].q[0] = *(const u32x4*)p;                                 \
            AA[mi].q[1] = *(const u32x4*)(p + 8);                           \
        }                                                                   \
        _Pragma("unroll")                                                   \
        for (int nj = 0; nj < 4; ++nj) {                                    \
            const unsigned short* pb = rowB[nj] + (kk) + khalf * 16;        \
            BB[nj].q[0] = *(const u32x4*)pb;                                \
            BB[nj].q[1] = *(const u32x4*)(pb + 8);                          \
        }                                                                   \
    } while (0)

#define FC2_MMA(AA, BB) do {                                                \
        _Pragma("unroll")                                                   \
        for (int nj = 0; nj < 4; ++nj) {                                    \
            _Pragma("unroll")                                               \
            for (int mi = 0; mi < 2; ++mi)                                  \
                c[mi][nj] = __builtin_amdgcn_wmma_f32_16x16x32_bf16(        \
                    false, AA[mi].v, false, BB[nj].v, (short)0,             \
                    c[mi][nj], false, false);                               \
        }                                                                   \
    } while (0)

    FC2_LOAD(a0, b0, 0);
    for (int k0 = 0; k0 < F; k0 += 64) {          // F multiple of 64
        __builtin_prefetch(rowB[0] + k0 + 64, 0, 1);
        FC2_LOAD(a1, b1, k0 + 32);
        FC2_MMA(a0, b0);
        if (k0 + 64 < F) FC2_LOAD(a0, b0, k0 + 64);
        FC2_MMA(a1, b1);
    }
#undef FC2_LOAD
#undef FC2_MMA

#pragma unroll
    for (int mi = 0; mi < 2; ++mi) {
        int mbase = s0 + mi * 16 + khalf * 8;
#pragma unroll
        for (int r = 0; r < 8; ++r) {
            int row = mbase + r;
            int tok = shared_mode ? row : slot_tok[row];
#pragma unroll
            for (int nj = 0; nj < 4; ++nj) {
                int n = n0 + nj * 16 + col;
                float v = c[mi][nj][r];
                if (shared_mode) {
                    out[(size_t)row * Dout + n] = v;       // base: shared expert
                } else if (tok >= 0) {
                    atomicAdd(&out[(size_t)tok * Dout + n], v);  // routed combine
                }
            }
        }
    }
}

// ---------------- host launch ----------------
extern "C" void kernel_launch(void* const* d_in, const int* in_sizes, int n_in,
                              void* d_out, int out_size, void* d_ws, size_t ws_size,
                              hipStream_t stream) {
    const float* x   = (const float*)d_in[0];
    const float* gw  = (const float*)d_in[1];
    const float* Wg  = (const float*)d_in[2];
    const float* Wu  = (const float*)d_in[3];
    const float* Wd  = (const float*)d_in[4];
    const float* Wgs = (const float*)d_in[5];
    const float* Wus = (const float*)d_in[6];
    const float* Wds = (const float*)d_in[7];
    const float* sgw = (const float*)d_in[8];
    float* out    = (float*)d_out;
    float* logits = out + (size_t)T_TOK * DMODEL;

    char* ws = (char*)d_ws;
    size_t o = 0;
    auto carve = [&](size_t bytes) -> void* {
        void* p = ws + o;
        o += (bytes + 255) & ~(size_t)255;
        return p;
    };
    unsigned short* xb    = (unsigned short*)carve((size_t)T_TOK * DMODEL * 2);
    unsigned short* wg_b  = (unsigned short*)carve((size_t)NEXP * FE * DMODEL * 2);
    unsigned short* wu_b  = (unsigned short*)carve((size_t)NEXP * FE * DMODEL * 2);
    unsigned short* wd_b  = (unsigned short*)carve((size_t)NEXP * DMODEL * FE * 2);
    unsigned short* wgs_b = (unsigned short*)carve((size_t)DFF * DMODEL * 2);
    unsigned short* wus_b = (unsigned short*)carve((size_t)DFF * DMODEL * 2);
    unsigned short* wds_b = (unsigned short*)carve((size_t)DMODEL * DFF * 2);
    unsigned short* hbuf  = (unsigned short*)carve((size_t)MAX_SLOTS * FE * 2); // >= T*DFF*2
    int*   slot_tok = (int*)carve((size_t)MAX_SLOTS * 4);
    float* slot_w   = (float*)carve((size_t)MAX_SLOTS * 4);
    int*   tki      = (int*)carve((size_t)T_TOK * 2 * 4);
    float* tkw      = (float*)carve((size_t)T_TOK * 2 * 4);
    float* sg       = (float*)carve((size_t)T_TOK * 4);
    int*   cnt      = (int*)carve(64);
    int*   offs     = (int*)carve(64);
    int*   cur      = (int*)carve(64);

    auto cvt = [&](const float* s, unsigned short* d, size_t n) {
        int blk = (int)((n / 4 + 255) / 256);
        cvt_kernel<<<blk, 256, 0, stream>>>(s, d, n);
    };
    cvt(x,   xb,    (size_t)T_TOK * DMODEL);
    cvt(Wg,  wg_b,  (size_t)NEXP * FE * DMODEL);
    cvt(Wu,  wu_b,  (size_t)NEXP * FE * DMODEL);
    cvt(Wd,  wd_b,  (size_t)NEXP * DMODEL * FE);
    cvt(Wgs, wgs_b, (size_t)DFF * DMODEL);
    cvt(Wus, wus_b, (size_t)DFF * DMODEL);
    cvt(Wds, wds_b, (size_t)DMODEL * DFF);

    init_kernel<<<(MAX_SLOTS + 255) / 256, 256, 0, stream>>>(slot_tok, slot_w, cnt);
    router_kernel<<<T_TOK / 8, 256, 0, stream>>>(x, gw, sgw, logits, tki, tkw, sg, cnt);
    offsets_kernel<<<1, 32, 0, stream>>>(cnt, offs, cur);
    fill_kernel<<<(T_TOK + 255) / 256, 256, 0, stream>>>(tki, tkw, cur, slot_tok, slot_w);

    // shared expert (sigmoid gate folded into h): writes `out` directly
    fc1_kernel<<<dim3(T_TOK / 32, DFF / 512), 256, 0, stream>>>(
        xb, wgs_b, wus_b, hbuf, nullptr, sg, nullptr, DFF, 1);
    fc2_kernel<<<dim3(T_TOK / 32, DMODEL / 512), 256, 0, stream>>>(
        hbuf, wds_b, out, nullptr, nullptr, DFF, 1);

    // routed experts (routing weight folded into h): atomic-add combine
    fc1_kernel<<<dim3(MAX_SLOTS / 32, FE / 512), 256, 0, stream>>>(
        xb, wg_b, wu_b, hbuf, slot_tok, slot_w, offs, FE, 0);
    fc2_kernel<<<dim3(MAX_SLOTS / 32, DMODEL / 512), 256, 0, stream>>>(
        hbuf, wd_b, out, slot_tok, offs, FE, 0);

    (void)in_sizes; (void)n_in; (void)out_size; (void)ws_size;
}
